// TreeLSTM_73950746902726
// MI455X (gfx1250) — matrix operand order repaired and checked by reading, not compiled, not run on hardware
//
#include <hip/hip_runtime.h>
#include <hip/hip_bf16.h>
#include <math.h>

typedef __bf16 bf16_t;
typedef __attribute__((ext_vector_type(16))) __bf16 v16bf;
typedef __attribute__((ext_vector_type(8)))  __bf16 v8bf;
typedef __attribute__((ext_vector_type(8)))  float  v8f;

#define TDEPTH  9
#define NTREES  128
#define NPT     1023
#define NNODES  (NTREES * NPT)     // 130944
#define HSZ     256
#define NLEAF   (NTREES * 512)     // 65536
#define VOCAB   32000

__device__ __forceinline__ float sigmf(float x) { return 1.0f / (1.0f + __expf(-x)); }

__device__ __forceinline__ v16bf cat8(v8bf lo, v8bf hi) {
    return __builtin_shufflevector(lo, hi, 0,1,2,3,4,5,6,7,8,9,10,11,12,13,14,15);
}

// ---------------- pack kernels ----------------

__global__ void k_pack_emb(const float* __restrict__ src, bf16_t* __restrict__ dst, int n) {
    int i = blockIdx.x * 256 + threadIdx.x;
    if (i < n) dst[i] = (bf16_t)src[i];
}

// Wcat = [U_f_w (512x512) | U_iou (512x768)]  -> packed: dst[((k>>5)*1280 + n)*32 + (k&31)]
__global__ void k_pack_wcat(const float* __restrict__ ufw, const float* __restrict__ uiou,
                            bf16_t* __restrict__ dst) {
    int i = blockIdx.x * 256 + threadIdx.x;       // 512*1280 = 655360 total
    int k = i / 1280;
    int n = i - k * 1280;
    float v = (n < 512) ? ufw[(size_t)k * 512 + n] : uiou[(size_t)k * 768 + (n - 512)];
    dst[((size_t)(k >> 5) * 1280 + n) * 32 + (k & 31)] = (bf16_t)v;
}

// W_iou (256x768) -> packed: dst[((k>>5)*768 + n)*32 + (k&31)]
__global__ void k_pack_wleaf(const float* __restrict__ wiou, bf16_t* __restrict__ dst) {
    int i = blockIdx.x * 256 + threadIdx.x;       // 256*768 = 196608 total
    int k = i / 768;
    int n = i - k * 768;
    dst[((size_t)(k >> 5) * 768 + n) * 32 + (k & 31)] = (bf16_t)wiou[(size_t)k * 768 + n];
}

// ---------------- leaf kernel: iou0 = (emb[wid*mask] @ W_iou), then gates ----------------
// Block: 512 thr = 16 waves (2 M-tiles x 8 N-chunks). Tile M=32 x N=768, K=256.

__global__ __launch_bounds__(512)
void k_leaf(const bf16_t* __restrict__ embb, const int* __restrict__ wordid,
            const int* __restrict__ maskp, const bf16_t* __restrict__ wpack,
            const float* __restrict__ biou,
            bf16_t* __restrict__ hb, float* __restrict__ cb, bf16_t* __restrict__ mhb) {
    __shared__ _Float16 smem[32 * 768];           // 48 KB

    const int tid  = threadIdx.x;
    const int lane = tid & 31;
    const int wave = tid >> 5;
    const int wm   = wave >> 3;                   // 0..1 -> M half
    const int wn   = wave & 7;                    // 0..7 -> N chunk of 96
    const int half = lane >> 4;
    const int lm   = lane & 15;
    const int mblk = blockIdx.x;

    // A row for this lane: leaf m -> global node g
    const int m = mblk * 32 + wm * 16 + lm;
    const size_t g = (size_t)(m >> 9) * NPT + 511 + (m & 511);
    const int wid = wordid[g] * maskp[g];
    const bf16_t* arow = embb + (size_t)wid * HSZ;

    v8f zero = {};
    v8f acc[6];
    #pragma unroll
    for (int t = 0; t < 6; ++t) acc[t] = zero;

    #pragma unroll 1
    for (int ks = 0; ks < 8; ++ks) {              // K = 256
        const bf16_t* asrc = arow + ks * 32 + half * 8;
        v16bf A = cat8(*(const v8bf*)asrc, *(const v8bf*)(asrc + 16));
        const bf16_t* bks = wpack + (size_t)ks * 768 * 32;
        // unconditional: last iteration prefetches into the next ws buffer (harmless)
        __builtin_prefetch(bks + (size_t)768 * 32 + (size_t)(wn * 96 + lm) * 32, 0, 1);
        #pragma unroll
        for (int t = 0; t < 6; ++t) {
            const int n = wn * 96 + t * 16 + lm;
            const bf16_t* bsrc = bks + (size_t)n * 32 + half * 8;
            v16bf B = cat8(*(const v8bf*)bsrc, *(const v8bf*)(bsrc + 16));
            acc[t] = __builtin_amdgcn_wmma_f32_16x16x32_bf16(
                false, A, false, B, (short)0, acc[t], false, false);
        }
    }

    // C/D layout: vgpr v, lanes0-15 -> M=v, lanes16-31 -> M=v+8, N=lane&15
    #pragma unroll
    for (int t = 0; t < 6; ++t) {
        const int n = wn * 96 + t * 16 + lm;
        #pragma unroll
        for (int v = 0; v < 8; ++v)
            smem[(size_t)(wm * 16 + v + 8 * half) * 768 + n] = (_Float16)acc[t][v];
    }
    __syncthreads();

    // epilogue: i,o,u -> h,c,mh (c_in=0, mh_in=0); iou scaled by mask BEFORE bias
    for (int idx = tid; idx < 32 * HSZ; idx += 512) {
        const int r = idx >> 8;
        const int d = idx & (HSZ - 1);
        const int mm = mblk * 32 + r;
        const size_t gg = (size_t)(mm >> 9) * NPT + 511 + (mm & 511);
        const float mk = (float)maskp[gg];
        const _Float16* row = smem + (size_t)r * 768;
        float iv = (float)row[d]       * mk + biou[d];
        float ov = (float)row[256 + d] * mk + biou[256 + d];
        float uv = (float)row[512 + d] * mk + biou[512 + d];
        float cv = sigmf(iv) * tanhf(uv);
        float hv = sigmf(ov) * tanhf(cv);
        float mv = fmaxf(hv, 0.0f);
        hb [gg * HSZ + d] = (bf16_t)hv;
        cb [gg * HSZ + d] = cv;
        mhb[gg * HSZ + d] = (bf16_t)mv;
    }
}

// ---------------- internal level kernel: h_cat @ [U_f_w | U_iou], then gates ----------------
// Block: 512 thr = 16 waves (2 M-tiles x 8 N-chunks). Tile M=32 x N=1280, K=512.

__global__ __launch_bounds__(512)
void k_level(const bf16_t* __restrict__ wpack,
             const float* __restrict__ ufb, const float* __restrict__ biou,
             bf16_t* __restrict__ hb, float* __restrict__ cb, bf16_t* __restrict__ mhb,
             int lvl) {
    __shared__ _Float16 smem[32 * 1280];          // 80 KB

    const int tid  = threadIdx.x;
    const int lane = tid & 31;
    const int wave = tid >> 5;
    const int wm   = wave >> 3;                   // 0..1 -> M half
    const int wn   = wave & 7;                    // 0..7 -> N chunk of 160
    const int half = lane >> 4;
    const int lm   = lane & 15;
    const int mblk = blockIdx.x;

    // A row: node m at this level; A[:,0:256]=h[left], A[:,256:512]=h[right]
    const int m    = mblk * 32 + wm * 16 + lm;
    const int lw   = 1 << lvl;
    const int tree = m >> lvl;
    const int j    = m & (lw - 1);
    const int loc  = (lw - 1) + j;
    const size_t clg = (size_t)tree * NPT + 2 * loc + 1;
    const bf16_t* hl = hb + clg * HSZ;
    const bf16_t* hr = hl + HSZ;                  // right child = left + 1

    v8f zero = {};
    v8f acc[10];
    #pragma unroll
    for (int t = 0; t < 10; ++t) acc[t] = zero;

    #pragma unroll 1
    for (int ks = 0; ks < 16; ++ks) {             // K = 512
        const int k0 = ks * 32;
        const bf16_t* asrc = (k0 < HSZ) ? (hl + k0 + half * 8)
                                        : (hr + (k0 - HSZ) + half * 8);
        v16bf A = cat8(*(const v8bf*)asrc, *(const v8bf*)(asrc + 16));
        const bf16_t* bks = wpack + (size_t)ks * 1280 * 32;
        // unconditional: last iteration prefetches into the next ws buffer (harmless)
        __builtin_prefetch(bks + (size_t)1280 * 32 + (size_t)(wn * 160 + lm) * 32, 0, 1);
        #pragma unroll
        for (int t = 0; t < 10; ++t) {
            const int n = wn * 160 + t * 16 + lm;
            const bf16_t* bsrc = bks + (size_t)n * 32 + half * 8;
            v16bf B = cat8(*(const v8bf*)bsrc, *(const v8bf*)(bsrc + 16));
            acc[t] = __builtin_amdgcn_wmma_f32_16x16x32_bf16(
                false, A, false, B, (short)0, acc[t], false, false);
        }
    }

    #pragma unroll
    for (int t = 0; t < 10; ++t) {
        const int n = wn * 160 + t * 16 + lm;
        #pragma unroll
        for (int v = 0; v < 8; ++v)
            smem[(size_t)(wm * 16 + v + 8 * half) * 1280 + n] = (_Float16)acc[t][v];
    }
    __syncthreads();

    // epilogue: cols [0,512)=f (bias U_f_b), [512,1280)=i,o,u (bias b_iou)
    for (int idx = tid; idx < 32 * HSZ; idx += 512) {
        const int r  = idx >> 8;
        const int d  = idx & (HSZ - 1);
        const int mm = mblk * 32 + r;
        const int tr = mm >> lvl;
        const int jj = mm & (lw - 1);
        const int lc = (lw - 1) + jj;
        const size_t gg = (size_t)tr * NPT + lc;
        const size_t c0 = (size_t)tr * NPT + 2 * lc + 1;   // left child; right = c0+1
        const _Float16* row = smem + (size_t)r * 1280;
        float fl = sigmf((float)row[d]        + ufb[d]);
        float fr = sigmf((float)row[256 + d]  + ufb[256 + d]);
        float iv = (float)row[512 + d]  + biou[d];
        float ov = (float)row[768 + d]  + biou[256 + d];
        float uv = (float)row[1024 + d] + biou[512 + d];
        float cv = sigmf(iv) * tanhf(uv)
                 + fl * cb[c0 * HSZ + d] + fr * cb[(c0 + 1) * HSZ + d];
        float hv = sigmf(ov) * tanhf(cv);
        float mv = fmaxf(hv, fmaxf((float)mhb[c0 * HSZ + d], (float)mhb[(c0 + 1) * HSZ + d]));
        hb [gg * HSZ + d] = (bf16_t)hv;
        cb [gg * HSZ + d] = cv;
        mhb[gg * HSZ + d] = (bf16_t)mv;
    }
}

// ---------------- output: logits = (h + mh) @ lin_w + lin_b ----------------

__global__ __launch_bounds__(128)
void k_out(const bf16_t* __restrict__ hb, const bf16_t* __restrict__ mhb,
           const float* __restrict__ lw, const float* __restrict__ lb,
           float* __restrict__ out) {
    const int g = blockIdx.x * 128 + threadIdx.x;     // 1023*128 == NNODES exactly
    float a0 = lb[0], a1 = lb[1], a2 = lb[2], a3 = lb[3];
    const bf16_t* hp = hb  + (size_t)g * HSZ;
    const bf16_t* mp = mhb + (size_t)g * HSZ;
    #pragma unroll 4
    for (int d = 0; d < HSZ; ++d) {
        float x = (float)hp[d] + (float)mp[d];
        a0 += x * lw[d * 4 + 0];
        a1 += x * lw[d * 4 + 1];
        a2 += x * lw[d * 4 + 2];
        a3 += x * lw[d * 4 + 3];
    }
    out[(size_t)g * 4 + 0] = a0;
    out[(size_t)g * 4 + 1] = a1;
    out[(size_t)g * 4 + 2] = a2;
    out[(size_t)g * 4 + 3] = a3;
}

// ---------------- host ----------------

extern "C" void kernel_launch(void* const* d_in, const int* in_sizes, int n_in,
                              void* d_out, int out_size, void* d_ws, size_t ws_size,
                              hipStream_t stream) {
    (void)in_sizes; (void)n_in; (void)out_size; (void)ws_size;
    const int*   wordid = (const int*)  d_in[0];
    const int*   maskp  = (const int*)  d_in[1];
    const float* emb    = (const float*)d_in[2];
    const float* Wiou   = (const float*)d_in[3];
    const float* Uiou   = (const float*)d_in[4];
    const float* Ufw    = (const float*)d_in[5];
    const float* Ufb    = (const float*)d_in[6];
    const float* biou   = (const float*)d_in[7];
    const float* linw   = (const float*)d_in[8];
    const float* linb   = (const float*)d_in[9];
    float* out = (float*)d_out;

    char* ws = (char*)d_ws;
    auto carve = [&](size_t bytes) -> char* {
        char* p = ws;
        ws += (bytes + 255) & ~(size_t)255;
        return p;
    };
    bf16_t* embb  = (bf16_t*)carve((size_t)VOCAB * HSZ * 2);     // 16.0 MB
    bf16_t* wcat  = (bf16_t*)carve((size_t)512 * 1280 * 2);      //  1.3 MB
    bf16_t* wleaf = (bf16_t*)carve((size_t)256 * 768 * 2);       //  0.4 MB
    bf16_t* hb    = (bf16_t*)carve((size_t)NNODES * HSZ * 2);    // 67.0 MB
    bf16_t* mhb   = (bf16_t*)carve((size_t)NNODES * HSZ * 2);    // 67.0 MB
    float*  cbuf  = (float*) carve((size_t)NNODES * HSZ * 4);    // 134.1 MB

    k_pack_emb  <<<VOCAB, 256, 0, stream>>>(emb, embb, VOCAB * HSZ);
    k_pack_wcat <<<2560, 256, 0, stream>>>(Ufw, Uiou, wcat);
    k_pack_wleaf<<<768,  256, 0, stream>>>(Wiou, wleaf);

    k_leaf<<<NLEAF / 32, 512, 0, stream>>>(embb, wordid, maskp, wleaf, biou, hb, cbuf, mhb);

    for (int l = TDEPTH - 1; l >= 0; --l) {
        int mblocks = (NTREES << l) / 32;   // 4<<l
        k_level<<<mblocks, 512, 0, stream>>>(wcat, Ufb, biou, hb, cbuf, mhb, l);
    }

    k_out<<<NNODES / 128, 128, 0, stream>>>(hb, mhb, linw, linb, out);
}
